// ATTCNN_84988812853674
// MI455X (gfx1250) — matrix operand order repaired and checked by reading, not compiled, not run on hardware
//
#include <hip/hip_runtime.h>
#include <hip/hip_bf16.h>
#include <math.h>

typedef _Float16 v16h __attribute__((ext_vector_type(16)));
typedef _Float16 v8h  __attribute__((ext_vector_type(8)));
typedef _Float16 v2h  __attribute__((ext_vector_type(2)));
typedef float    v8f  __attribute__((ext_vector_type(8)));

#define B_    512
#define L_    128
#define DIM_  100
#define PDIM_ 25
#define FEAT_ 350
#define FEATP_ 352
#define FNUM_ 256
#define LAB_  19
#define KCH1  11   // 352/32 K-chunks for conv GEMM
#define KCH2  8    // 256/32 K-chunks for attention GEMM

// ws layout: packed conv weights (f16 fragments), then packed U@WL (f16 fragments)
#define WSW_ELEMS  (3*KCH1*16*512)   // 270336 f16
#define WSW_BYTES  (WSW_ELEMS*2)     // 540672 B
#define WSUW_ELEMS (KCH2*16*512)     // 65536 f16

// LDS byte offsets (G overlays wePad+Rpad; both dead before G is written)
#define OFF_WEPAD 0
#define OFF_G     0
#define OFF_RPAD  52000
#define OFF_RCT   143520
#define OFF_S1    209056
#define OFF_S2    209568
#define OFF_INE   210080
#define OFF_L1    210592
#define OFF_L2    210992
#define OFF_RED   211392
#define SMEM_BYTES 212480

__device__ __forceinline__ float fast_tanh(float x) {
#if __has_builtin(__builtin_amdgcn_tanhf)
  return __builtin_amdgcn_tanhf(x);        // gfx1250 v_tanh_f32
#else
  float e = __expf(2.f * x);               // branchless, exact at +/-inf
  return 1.f - 2.f / (e + 1.f);
#endif
}

__device__ __forceinline__ v16h ld_frag_lds(const _Float16* p) {
  v8h lo = *(const v8h*)p;
  v8h hi = *(const v8h*)(p + 8);
  return __builtin_shufflevector(lo, hi, 0,1,2,3,4,5,6,7,8,9,10,11,12,13,14,15);
}
__device__ __forceinline__ v16h ld_frag_g(const _Float16* p) {
  v8h lo = *(const v8h*)p;
  v8h hi = *(const v8h*)(p + 8);
  return __builtin_shufflevector(lo, hi, 0,1,2,3,4,5,6,7,8,9,10,11,12,13,14,15);
}

// Pack conv_w (FNUM,1,3,FEAT) f32 -> f16 WMMA B-fragments.
// Fragment order: (((k*11+chunk)*16+ntile)*32+lane)*16+t
// lane<16 holds K=chunk*32+0..15, lane>=16 holds K=chunk*32+16..31; col = ntile*16+(lane&15)
__global__ void attcnn_pack_w(const float* __restrict__ convw, _Float16* __restrict__ wsW) {
  int idx = blockIdx.x * 256 + threadIdx.x;
  if (idx >= WSW_ELEMS) return;
  int t     = idx & 15;
  int lane  = (idx >> 4) & 31;
  int ntile = (idx >> 9) & 15;
  int chunk = (idx >> 13) % 11;
  int k     = idx / (512 * 16 * 11);
  int d = chunk * 32 + ((lane & 16) ? 16 : 0) + t;
  int f = ntile * 16 + (lane & 15);
  float v = (d < FEAT_) ? convw[f * (3 * FEAT_) + k * FEAT_ + d] : 0.f;
  wsW[idx] = (_Float16)v;
}

// UW = U(256x19) @ WL(19x256), packed into f16 B-fragments: ((chunk*16+ntile)*32+lane)*16+t
__global__ void attcnn_pack_uw(const float* __restrict__ U, const float* __restrict__ WL,
                               _Float16* __restrict__ wsUW) {
  int idx = blockIdx.x * 256 + threadIdx.x;
  if (idx >= WSUW_ELEMS) return;
  int t     = idx & 15;
  int lane  = (idx >> 4) & 31;
  int ntile = (idx >> 9) & 15;
  int chunk = idx >> 13;                       // 0..7
  int kf = chunk * 32 + ((lane & 16) ? 16 : 0) + t;  // f index
  int g  = ntile * 16 + (lane & 15);
  float a = 0.f;
  #pragma unroll
  for (int c = 0; c < LAB_; ++c) a += U[kf * LAB_ + c] * WL[c * FNUM_ + g];
  wsUW[idx] = (_Float16)a;
}

// R value at conv-input row r (0..129), feature column c (0..351)
__device__ __forceinline__ float rpad_val(int r, int c, const float* wePad, const float* ine,
                                          const int* p1, const int* p2,
                                          const float* pos1, const float* pos2, int b) {
  if (r < 1 || r > L_ || c >= FEAT_) return 0.f;   // conv halo rows + K-pad cols
  int l = r - 1;
  float val;
  if (c < 3 * DIM_) {
    val = (l == 0) ? 0.f : wePad[(l + c / DIM_) * DIM_ + (c % DIM_)];  // WF[:,0]=0
  } else if (c < 3 * DIM_ + PDIM_) {
    val = pos1[p1[b * L_ + l] * PDIM_ + (c - 3 * DIM_)];
  } else {
    val = pos2[p2[b * L_ + l] * PDIM_ + (c - 3 * DIM_ - PDIM_)];
  }
  return val * ine[l];
}

__global__ __launch_bounds__(256, 1)
void attcnn_main(const int* __restrict__ inputs, const int* __restrict__ e1s,
                 const int* __restrict__ e1e, const int* __restrict__ e2s,
                 const int* __restrict__ e2e, const int* __restrict__ p1,
                 const int* __restrict__ p2, const float* __restrict__ emb,
                 const float* __restrict__ pos1, const float* __restrict__ pos2,
                 const float* __restrict__ convb, const float* __restrict__ WL,
                 const _Float16* __restrict__ wsW, const _Float16* __restrict__ wsUW,
                 float* __restrict__ out) {
  extern __shared__ char smem[];
  float*    wePad = (float*)(smem + OFF_WEPAD);
  _Float16* rpad  = (_Float16*)(smem + OFF_RPAD);
  _Float16* rct   = (_Float16*)(smem + OFF_RCT);
  float*    Gm    = (float*)(smem + OFF_G);
  float*    s1a   = (float*)(smem + OFF_S1);
  float*    s2a   = (float*)(smem + OFF_S2);
  float*    ine   = (float*)(smem + OFF_INE);
  float*    l1a   = (float*)(smem + OFF_L1);
  float*    l2a   = (float*)(smem + OFF_L2);
  float*    red   = (float*)(smem + OFF_RED);

  const int b    = blockIdx.x;
  const int tid  = threadIdx.x;
  const int lane = tid & 31;
  const int wv   = tid >> 5;
  const int mh   = wv >> 2;   // 0..1: row half (64 rows)
  const int nq   = wv & 3;    // 0..3: col quarter (64 cols)

  // ---- Phase 1a: gather padded word embeddings as float4 (rows 0,129 = pad token 0)
  {
    const float4* embv = (const float4*)emb;     // 100 floats = 25 float4 per row
    float4* weV = (float4*)wePad;
    for (int idx = tid; idx < 130 * 25; idx += 256) {
      int r = idx / 25, q = idx - r * 25;
      int tok = (r == 0 || r == 129) ? 0 : inputs[b * L_ + (r - 1)];
      weV[idx] = embv[tok * 25 + q];
    }
  }
  __syncthreads();

  // ---- Phase 1b: entity span means ----
  if (tid < DIM_) {
    int s = e1s[b], e = e1e[b];
    float a = 0.f;
    for (int l = s; l <= e; ++l) a += wePad[(l + 1) * DIM_ + tid];
    l1a[tid] = a / (float)(e - s + 1);
  } else if (tid < 2 * DIM_) {
    int d = tid - DIM_;
    int s = e2s[b], e = e2e[b];
    float a = 0.f;
    for (int l = s; l <= e; ++l) a += wePad[(l + 1) * DIM_ + d];
    l2a[d] = a / (float)(e - s + 1);
  }
  __syncthreads();

  // ---- Phase 1c: attention logits s1/s2 (float4 dot over DIM) ----
  {
    int l = (tid < L_) ? tid : tid - L_;
    const float4* wr = (const float4*)(wePad + (l + 1) * DIM_);
    const float4* lv = (const float4*)((tid < L_) ? l1a : l2a);
    float a = 0.f;
    #pragma unroll 5
    for (int q = 0; q < 25; ++q) {
      float4 w = wr[q], x = lv[q];
      a += w.x * x.x + w.y * x.y + w.z * x.z + w.w * x.w;
    }
    if (tid < L_) s1a[l] = a; else s2a[l] = a;
  }
  __syncthreads();

  // ---- dual softmax over L via LDS tree (both halves in parallel) ----
  float myv = (tid < L_) ? s1a[tid] : s2a[tid - L_];
  red[tid] = myv;
  __syncthreads();
  for (int off = 64; off; off >>= 1) {
    if ((tid & 127) < off) red[tid] = fmaxf(red[tid], red[tid + off]);
    __syncthreads();
  }
  float mx = (tid < L_) ? red[0] : red[L_];
  __syncthreads();
  float ev = __expf(myv - mx);
  red[tid] = ev;
  if (tid < L_) s1a[tid] = ev; else s2a[tid - L_] = ev;
  __syncthreads();
  for (int off = 64; off; off >>= 1) {
    if ((tid & 127) < off) red[tid] += red[tid + off];
    __syncthreads();
  }
  float sum1 = red[0], sum2 = red[L_];
  __syncthreads();
  if (tid < L_) ine[tid] = 0.5f * (s1a[tid] / sum1 + s2a[tid] / sum2);
  __syncthreads();

  // ---- Phase 1d: build f16 conv input Rpad[130][352] as f16 pairs ----
  for (int idx = tid; idx < 130 * (FEATP_ / 2); idx += 256) {
    int r = idx / (FEATP_ / 2);
    int c = (idx - r * (FEATP_ / 2)) * 2;
    v2h pr;
    pr.x = (_Float16)rpad_val(r, c,     wePad, ine, p1, p2, pos1, pos2, b);
    pr.y = (_Float16)rpad_val(r, c + 1, wePad, ine, p1, p2, pos1, pos2, b);
    *(v2h*)(rpad + r * FEATP_ + c) = pr;   // one ds_store_b32 per pair
  }
  __syncthreads();

  const int kb = (lane < 16) ? 0 : 16;   // A-fragment K sub-base (contiguous k = 16h + s, same as B packing)
  const v8f vzero = {0.f, 0.f, 0.f, 0.f, 0.f, 0.f, 0.f, 0.f};

  // ---- Phase B: conv GEMM (M=128, N=256, K=3*352) with v_wmma_f32_16x16x32_f16 ----
  {
    v8f acc[4][4];
    #pragma unroll
    for (int mt = 0; mt < 4; ++mt)
      #pragma unroll
      for (int nt = 0; nt < 4; ++nt) acc[mt][nt] = vzero;

    #pragma unroll 1
    for (int k = 0; k < 3; ++k) {
      #pragma unroll 1
      for (int ch = 0; ch < KCH1; ++ch) {
        int col0 = ch * 32 + kb;
        v16h aF[4];
        #pragma unroll
        for (int mt = 0; mt < 4; ++mt) {
          int row = mh * 64 + mt * 16 + (lane & 15) + k;   // Rpad row = l + k
          aF[mt] = ld_frag_lds(rpad + row * FEATP_ + col0);
        }
        #pragma unroll
        for (int nt = 0; nt < 4; ++nt) {
          int ntg = nq * 4 + nt;
          v16h bF = ld_frag_g(wsW + (size_t)(((k * KCH1 + ch) * 16 + ntg) * 32 + lane) * 16);
          #pragma unroll
          for (int mt = 0; mt < 4; ++mt)
            acc[mt][nt] = __builtin_amdgcn_wmma_f32_16x16x32_f16(
                false, aF[mt], false, bF, (short)0, acc[mt][nt], false, false);
        }
      }
    }
    // epilogue: bias + tanh -> RcT[l][f] (f16), hardware v_tanh path
    #pragma unroll
    for (int nt = 0; nt < 4; ++nt) {
      int f = nq * 64 + nt * 16 + (lane & 15);
      float bias = convb[f];
      #pragma unroll
      for (int mt = 0; mt < 4; ++mt) {
        int lbase = mh * 64 + mt * 16 + ((lane < 16) ? 0 : 8);
        #pragma unroll
        for (int r = 0; r < 8; ++r) {
          float v = fast_tanh(acc[mt][nt][r] + bias);
          rct[(lbase + r) * FNUM_ + f] = (_Float16)v;
        }
      }
    }
  }
  __syncthreads();

  // ---- Phase C: G = RcT(128x256) @ UW(256x256) via WMMA ----
  {
    v8f acc[4][4];
    #pragma unroll
    for (int mt = 0; mt < 4; ++mt)
      #pragma unroll
      for (int nt = 0; nt < 4; ++nt) acc[mt][nt] = vzero;

    #pragma unroll 1
    for (int ch = 0; ch < KCH2; ++ch) {
      int col0 = ch * 32 + kb;
      v16h aF[4];
      #pragma unroll
      for (int mt = 0; mt < 4; ++mt) {
        int row = mh * 64 + mt * 16 + (lane & 15);
        aF[mt] = ld_frag_lds(rct + row * FNUM_ + col0);
      }
      #pragma unroll
      for (int nt = 0; nt < 4; ++nt) {
        int ntg = nq * 4 + nt;
        v16h bF = ld_frag_g(wsUW + (size_t)((ch * 16 + ntg) * 32 + lane) * 16);
        #pragma unroll
        for (int mt = 0; mt < 4; ++mt)
          acc[mt][nt] = __builtin_amdgcn_wmma_f32_16x16x32_f16(
              false, aF[mt], false, bF, (short)0, acc[mt][nt], false, false);
      }
    }
    // write G f32 into LDS (overlays dead wePad/Rpad region)
    #pragma unroll
    for (int nt = 0; nt < 4; ++nt) {
      int g = nq * 64 + nt * 16 + (lane & 15);
      #pragma unroll
      for (int mt = 0; mt < 4; ++mt) {
        int lbase = mh * 64 + mt * 16 + ((lane < 16) ? 0 : 8);
        #pragma unroll
        for (int r = 0; r < 8; ++r) Gm[(lbase + r) * FNUM_ + g] = acc[mt][nt][r];
      }
    }
  }
  __syncthreads();

  // ---- Phase D: softmax over L per column g, then wo[f] = max_l Rc*G_ ----
  {
    int g = tid;                       // 256 threads = 256 columns
    float m = -3.402823e38f;
    for (int l = 0; l < L_; ++l) m = fmaxf(m, Gm[l * FNUM_ + g]);
    float s = 0.f;
    for (int l = 0; l < L_; ++l) {
      float e = __expf(Gm[l * FNUM_ + g] - m);
      Gm[l * FNUM_ + g] = e;           // thread owns its column: no race
      s += e;
    }
    float inv = 1.f / s;
    float wo = -3.402823e38f;
    for (int l = 0; l < L_; ++l)
      wo = fmaxf(wo, (float)rct[l * FNUM_ + g] * Gm[l * FNUM_ + g] * inv);
    out[b * FNUM_ + g] = wo;
  }

  // second output: WL passthrough (block 0 only)
  if (b == 0) {
    for (int i = tid; i < LAB_ * FNUM_; i += 256) out[B_ * FNUM_ + i] = WL[i];
  }
}

extern "C" void kernel_launch(void* const* d_in, const int* in_sizes, int n_in,
                              void* d_out, int out_size, void* d_ws, size_t ws_size,
                              hipStream_t stream) {
  (void)in_sizes; (void)n_in; (void)out_size; (void)ws_size;
  const int*   inputs = (const int*)d_in[0];
  const int*   e1s    = (const int*)d_in[1];
  const int*   e1e    = (const int*)d_in[2];
  const int*   e2s    = (const int*)d_in[3];
  const int*   e2e    = (const int*)d_in[4];
  const int*   p1     = (const int*)d_in[5];
  const int*   p2     = (const int*)d_in[6];
  const float* emb    = (const float*)d_in[7];
  const float* pos1   = (const float*)d_in[8];
  const float* pos2   = (const float*)d_in[9];
  const float* convw  = (const float*)d_in[10];
  const float* convb  = (const float*)d_in[11];
  const float* U      = (const float*)d_in[12];
  const float* WL     = (const float*)d_in[13];
  float* out = (float*)d_out;
  _Float16* wsW  = (_Float16*)d_ws;
  _Float16* wsUW = (_Float16*)((char*)d_ws + WSW_BYTES);

  attcnn_pack_w<<<(WSW_ELEMS + 255) / 256, 256, 0, stream>>>(convw, wsW);
  attcnn_pack_uw<<<WSUW_ELEMS / 256, 256, 0, stream>>>(U, WL, wsUW);
  attcnn_main<<<B_, 256, SMEM_BYTES, stream>>>(inputs, e1s, e1e, e2s, e2e, p1, p2,
                                               emb, pos1, pos2, convb, WL,
                                               wsW, wsUW, out);
}
